// AttentionModern_84945863180967
// MI455X (gfx1250) — matrix operand order repaired and checked by reading, not compiled, not run on hardware
//
#include <hip/hip_runtime.h>

// ---------------------------------------------------------------------------
// Fused QKV projection: out = split3( hidden @ qkv_proj^T )
//   A = hidden_states  [M=16384, K=4096] f32   W = qkv_proj [N=12288, K=4096] f32
//   C[m,n] = sum_k A[m,k] * W[n,k];  d_out = [Q|K|V] thirds along n.
// Compute-bound (AI ~1350 FLOP/B) -> bf16 WMMA, f32 accumulate.
// Fast path: one-time f32->bf16 conversion into d_ws, then a double-buffered
// GEMM whose LDS staging uses CDNA5 async-to-LDS DMA (ASYNCcnt).
// Fallback (small ws): round-1 kernel with in-loop conversion.
// ---------------------------------------------------------------------------

typedef __bf16 bf16;
typedef __attribute__((ext_vector_type(16))) __bf16 v16bf;
typedef __attribute__((ext_vector_type(8)))  __bf16 v8bf;   // 16 bytes
typedef __attribute__((ext_vector_type(4)))  __bf16 v4bf;   // 8 bytes
typedef __attribute__((ext_vector_type(8)))  float  v8f;

#define M_DIM 16384
#define K_DIM 4096
#define N_DIM 12288
#define E_DIM 4096
#define BM 128
#define BN 128
#define BK 64
#define BKP 72                  // padded LDS row stride in bf16 elems
#define ROWB (BKP * 2)          // 144 bytes/row: 16B aligned, bank-spread
#define HBUF ((BM + BN) * BKP)  // elems per ping-pong buffer (36,864 B)

union Frag16 { v16bf v; v8bf h[2]; };

// ---------------------------------------------------------------------------
// Pass 1: streaming f32 -> bf16 (n is a multiple of 2048 for both tensors)
// ---------------------------------------------------------------------------
__global__ __launch_bounds__(256)
void cvt_f32_to_bf16(const float* __restrict__ src, bf16* __restrict__ dst)
{
    const size_t i = ((size_t)blockIdx.x * 256 + threadIdx.x) * 8;
    const float4 a = *(const float4*)(src + i);
    const float4 b = *(const float4*)(src + i + 4);
    v8bf p;
    p[0] = (bf16)a.x; p[1] = (bf16)a.y; p[2] = (bf16)a.z; p[3] = (bf16)a.w;
    p[4] = (bf16)b.x; p[5] = (bf16)b.y; p[6] = (bf16)b.z; p[7] = (bf16)b.w;
    *(v8bf*)(dst + i) = p;
}

// ---------------------------------------------------------------------------
// Fast path: bf16 GEMM, ping-pong LDS fed by global_load_async_to_lds_b128
// ---------------------------------------------------------------------------
__global__ __launch_bounds__(256, 2)
void qkv_gemm_wmma_async(const bf16* __restrict__ A,
                         const bf16* __restrict__ W,
                         float* __restrict__ out)
{
    __shared__ bf16 smem[2 * HBUF];

    const int tid  = threadIdx.x;
    const int m0   = blockIdx.y * BM;
    const int n0   = blockIdx.x * BN;
    const int wave = tid >> 5;
    const int lane = tid & 31;
    const int half = lane >> 4;
    const int l16  = lane & 15;
    const int wm   = (wave & 3) << 5;   // wave M offset: 0,32,64,96
    const int wn   = (wave >> 2) << 6;  // wave N offset: 0,64

    // Per-thread async chunk mapping: tile = 128 rows x 128B = 1024 x 16B
    // chunks per matrix; 4 chunks per thread per matrix per K-iteration.
    uint32_t gOffA[4], gOffB[4], lOffA[4], lOffB[4];
    const uint32_t ldsBase = (uint32_t)(size_t)(void*)smem;  // AS3 offset
    #pragma unroll
    for (int q = 0; q < 4; ++q) {
        const int idx = tid + (q << 8);       // 0..1023
        const int row = idx >> 3;             // 8 chunks per 64-elem row
        const int c   = idx & 7;
        gOffA[q] = (uint32_t)((m0 + row) * (K_DIM * 2) + c * 16);
        gOffB[q] = (uint32_t)((n0 + row) * (K_DIM * 2) + c * 16);
        lOffA[q] = ldsBase + (uint32_t)(row * ROWB + c * 16);
        lOffB[q] = ldsBase + (uint32_t)(BM * ROWB + row * ROWB + c * 16);
    }

    // DMA one BK-slice of A and W into ping-pong buffer `buf`.
    auto issue = [&](int buf, int k0) {
        const uint32_t kbyte = (uint32_t)(k0 * 2);
        const uint32_t lbump = (uint32_t)(buf * (HBUF * 2));
        #pragma unroll
        for (int q = 0; q < 4; ++q) {
            asm volatile("global_load_async_to_lds_b128 %0, %1, %2"
                         :: "v"(lOffA[q] + lbump), "v"(gOffA[q] + kbyte), "s"(A)
                         : "memory");
            asm volatile("global_load_async_to_lds_b128 %0, %1, %2"
                         :: "v"(lOffB[q] + lbump), "v"(gOffB[q] + kbyte), "s"(W)
                         : "memory");
        }
    };

    v8f acc[2][4] = {};

    issue(0, 0);
    asm volatile("s_wait_asynccnt 0x0" ::: "memory");
    __syncthreads();

    const int NIT = K_DIM / BK;  // 64
    for (int kb = 0; kb < NIT; ++kb) {
        const int cur = kb & 1;
        if (kb + 1 < NIT) issue(cur ^ 1, (kb + 1) * BK);  // prefetch next slice

        const bf16* lA = smem + cur * HBUF;
        const bf16* lB = lA + BM * BKP;
        #pragma unroll
        for (int s = 0; s < 2; ++s) {
            const int kbk = s << 5;  // 0, 32
            Frag16 a[2], b[4];
            // A 16x32 frag: row=l16, K = kbk+half*8+{0..7} | kbk+16+half*8+{0..7}
            #pragma unroll
            for (int i = 0; i < 2; ++i) {
                const bf16* p = lA + (wm + (i << 4) + l16) * BKP + kbk + (half << 3);
                a[i].h[0] = *(const v8bf*)(p);
                a[i].h[1] = *(const v8bf*)(p + 16);
            }
            // B 32x16 frag: col=l16, K = kbk+half*8+{0..7} | kbk+16+half*8+{0..7}
            #pragma unroll
            for (int j = 0; j < 4; ++j) {
                const bf16* p = lB + (wn + (j << 4) + l16) * BKP + kbk + (half << 3);
                b[j].h[0] = *(const v8bf*)(p);
                b[j].h[1] = *(const v8bf*)(p + 16);
            }
            #pragma unroll
            for (int i = 0; i < 2; ++i)
                #pragma unroll
                for (int j = 0; j < 4; ++j)
                    acc[i][j] = __builtin_amdgcn_wmma_f32_16x16x32_bf16(
                        false, a[i].v, false, b[j].v, (short)0, acc[i][j],
                        false, false);
        }
        asm volatile("s_wait_asynccnt 0x0" ::: "memory");
        __syncthreads();
    }

    const size_t SEC = (size_t)M_DIM * E_DIM;
    #pragma unroll
    for (int i = 0; i < 2; ++i) {
        #pragma unroll
        for (int j = 0; j < 4; ++j) {
            const int gn = n0 + wn + (j << 4) + l16;
            const size_t base = (size_t)(gn >> 12) * SEC + (size_t)(gn & (E_DIM - 1));
            const int gm0 = m0 + wm + (i << 4) + (half << 3);
            #pragma unroll
            for (int r = 0; r < 8; ++r)
                __builtin_nontemporal_store(acc[i][j][r],
                    out + base + (size_t)(gm0 + r) * E_DIM);
        }
    }
}

// ---------------------------------------------------------------------------
// Fallback (ws too small): round-1 kernel, f32 inputs converted in-loop
// ---------------------------------------------------------------------------
__global__ __launch_bounds__(256, 2)
void qkv_gemm_wmma(const float* __restrict__ A,
                   const float* __restrict__ W,
                   float* __restrict__ out)
{
    __shared__ bf16 lA[BM * BKP];
    __shared__ bf16 lB[BN * BKP];

    const int tid  = threadIdx.x;
    const int m0   = blockIdx.y * BM;
    const int n0   = blockIdx.x * BN;
    const int wave = tid >> 5;
    const int lane = tid & 31;
    const int half = lane >> 4;
    const int l16  = lane & 15;
    const int wm   = (wave & 3) << 5;
    const int wn   = (wave >> 2) << 6;

    v8f acc[2][4] = {};

    for (int k0 = 0; k0 < K_DIM; k0 += BK) {
        #pragma unroll
        for (int q = 0; q < 8; ++q) {
            const int idx = tid + (q << 8);
            const int row = idx >> 4;
            const int c4  = (idx & 15) << 2;
            const float4 av = *(const float4*)(A + (size_t)(m0 + row) * K_DIM + k0 + c4);
            v4bf ap;
            ap.x = (bf16)av.x; ap.y = (bf16)av.y; ap.z = (bf16)av.z; ap.w = (bf16)av.w;
            *(v4bf*)(lA + row * BKP + c4) = ap;
            const float4 bv = *(const float4*)(W + (size_t)(n0 + row) * K_DIM + k0 + c4);
            v4bf bp;
            bp.x = (bf16)bv.x; bp.y = (bf16)bv.y; bp.z = (bf16)bv.z; bp.w = (bf16)bv.w;
            *(v4bf*)(lB + row * BKP + c4) = bp;
        }
        __syncthreads();

        #pragma unroll
        for (int s = 0; s < 2; ++s) {
            const int kb = s << 5;
            Frag16 a[2], b[4];
            #pragma unroll
            for (int i = 0; i < 2; ++i) {
                const bf16* p = lA + (wm + (i << 4) + l16) * BKP + kb + (half << 3);
                a[i].h[0] = *(const v8bf*)(p);
                a[i].h[1] = *(const v8bf*)(p + 16);
            }
            #pragma unroll
            for (int j = 0; j < 4; ++j) {
                const bf16* p = lB + (wn + (j << 4) + l16) * BKP + kb + (half << 3);
                b[j].h[0] = *(const v8bf*)(p);
                b[j].h[1] = *(const v8bf*)(p + 16);
            }
            #pragma unroll
            for (int i = 0; i < 2; ++i)
                #pragma unroll
                for (int j = 0; j < 4; ++j)
                    acc[i][j] = __builtin_amdgcn_wmma_f32_16x16x32_bf16(
                        false, a[i].v, false, b[j].v, (short)0, acc[i][j],
                        false, false);
        }
        __syncthreads();
    }

    const size_t SEC = (size_t)M_DIM * E_DIM;
    #pragma unroll
    for (int i = 0; i < 2; ++i) {
        #pragma unroll
        for (int j = 0; j < 4; ++j) {
            const int gn = n0 + wn + (j << 4) + l16;
            const size_t base = (size_t)(gn >> 12) * SEC + (size_t)(gn & (E_DIM - 1));
            const int gm0 = m0 + wm + (i << 4) + (half << 3);
            #pragma unroll
            for (int r = 0; r < 8; ++r)
                __builtin_nontemporal_store(acc[i][j][r],
                    out + base + (size_t)(gm0 + r) * E_DIM);
        }
    }
}

extern "C" void kernel_launch(void* const* d_in, const int* in_sizes, int n_in,
                              void* d_out, int out_size, void* d_ws, size_t ws_size,
                              hipStream_t stream) {
    (void)in_sizes; (void)n_in; (void)out_size;
    const float* hs = (const float*)d_in[0];   // hidden_states [16384,4096] f32
    const float* w  = (const float*)d_in[1];   // qkv_proj      [12288,4096] f32
    // d_in[2] = position_ids: unused (rotary embedding is an identity stub)
    float* out = (float*)d_out;

    const size_t ABYTES = (size_t)M_DIM * K_DIM * sizeof(bf16);  // 134,217,728
    const size_t WBYTES = (size_t)N_DIM * K_DIM * sizeof(bf16);  // 100,663,296
    dim3 grid(N_DIM / BN, M_DIM / BM);                           // 96 x 128

    if (ws_size >= ABYTES + WBYTES) {
        bf16* Abf = (bf16*)d_ws;
        bf16* Wbf = (bf16*)((char*)d_ws + ABYTES);
        cvt_f32_to_bf16<<<((size_t)M_DIM * K_DIM) / 2048, 256, 0, stream>>>(hs, Abf);
        cvt_f32_to_bf16<<<((size_t)N_DIM * K_DIM) / 2048, 256, 0, stream>>>(w, Wbf);
        qkv_gemm_wmma_async<<<grid, dim3(256), 0, stream>>>(Abf, Wbf, out);
    } else {
        qkv_gemm_wmma<<<grid, dim3(256), 0, stream>>>(hs, w, out);
    }
}
